// PVRNNLayer_7258494730550
// MI455X (gfx1250) — matrix operand order repaired and checked by reading, not compiled, not run on hardware
//
#include <hip/hip_runtime.h>

// ---------------- problem constants ----------------
#define D_   256
#define Z_   32
#define IN_  256
#define B_   128
#define T_   1000
#define EPS_ 1e-6f
#define W_   0.001f
#define W1_  0.001f
#define WD1_ 0.001f

typedef __attribute__((ext_vector_type(16))) _Float16 v16h;
typedef __attribute__((ext_vector_type(8)))  _Float16 v8h;
typedef __attribute__((ext_vector_type(8)))  float    v8f;

// padded LDS strides (bank-conflict avoidance; all keep 16B alignment)
#define DP   264   // d_f16 row stride (halves)  -> 528 B
#define ZP   40    // z_f16 row stride (halves)  -> 80 B
#define DZP  68    // dz row stride (floats)

// ---------------- WMMA fragment helpers ----------------
// B fragment stored pre-swizzled: 512 halves (1 KB) per 32x16 tile,
// 16 contiguous halves per lane (lane = n + 16*(k>=16), half = k&15).
__device__ __forceinline__ void store_b_swz(_Float16* table, int frag_idx,
                                            int kloc, int nloc, float w) {
    int lane = nloc + ((kloc & 16) ? 16 : 0);
    int half = kloc & 15;
    table[(size_t)frag_idx * 512 + lane * 16 + half] = (_Float16)w;
}

__device__ __forceinline__ v16h ld_b_frag(const _Float16* table, int frag_idx) {
    int lane = threadIdx.x & 31;
    const _Float16* p = table + (size_t)frag_idx * 512 + lane * 16;
    v8h lo = *(const v8h*)(p);       // K pairs -> VGPR 0..3
    v8h hi = *(const v8h*)(p + 8);   // K pairs -> VGPR 4..7
    return __builtin_shufflevector(lo, hi, 0,1,2,3,4,5,6,7,8,9,10,11,12,13,14,15);
}

// A fragment (16x32 f16) from plain row-major f16 LDS.
// lanes 0-15: row m0+L, K [k0,k0+16)
// lanes 16-31: row m0+L-16, K [k0+16,k0+32)
__device__ __forceinline__ v16h ld_a_frag(const _Float16* base, int stride, int m0, int k0) {
    int lane = threadIdx.x & 31;
    int row  = m0 + (lane & 15);
    int kk   = k0 + ((lane & 16) ? 16 : 0);
    const _Float16* p = base + row * stride + kk;
    v8h lo = *(const v8h*)(p);
    v8h hi = *(const v8h*)(p + 8);
    return __builtin_shufflevector(lo, hi, 0,1,2,3,4,5,6,7,8,9,10,11,12,13,14,15);
}

__device__ __forceinline__ v8f wmma_f16(v16h a, v16h b, v8f c) {
    return __builtin_amdgcn_wmma_f32_16x16x32_f16(false, a, false, b, (short)0, c, false, false);
}

// ================= Phase 1: X[b,t,:] = hd[b,t,:] @ Whd2h^T + bias_d =================
// One big GEMM: rows R = B*T = 128000, N = 256, K = 256. 128 rows per block.
#define P1_SMEM (131072 + 128 * DP * 2)   // Whd2h swz (128 KB) + A tile (66 KB)

__global__ __launch_bounds__(1024, 1)
void pvrnn_phase1(const float* __restrict__ hd, const float* __restrict__ Whd2h,
                  const float* __restrict__ bias_d, float* __restrict__ out) {
    extern __shared__ char smem[];
    _Float16* bsw  = (_Float16*)smem;              // 16 nt x 8 kt fragments
    _Float16* alds = (_Float16*)(smem + 131072);   // 128 x DP
    const int tid = threadIdx.x, lane = tid & 31, wave = tid >> 5;
    const size_t row0 = (size_t)blockIdx.x * 128;

    // B[k][n] = Whd2h[n][k] ; element e -> n=e/256, k=e%256 (coalesced in k)
    for (int e = tid; e < D_ * IN_; e += 1024) {
        int n = e >> 8, k = e & 255;
        store_b_swz(bsw, (n >> 4) * 8 + (k >> 5), k & 31, n & 15, Whd2h[e]);
    }
    for (int e = tid; e < 128 * IN_; e += 1024) {
        int r = e >> 8, k = e & 255;
        alds[r * DP + k] = (_Float16)hd[(row0 + r) * IN_ + k];
    }
    __syncthreads();

    const int mw = wave >> 2, nq = wave & 3;       // 8 M-tiles x 4 wave-cols
    v8f acc[4] = {};
#pragma unroll
    for (int kt = 0; kt < 8; ++kt) {
        v16h a = ld_a_frag(alds, DP, mw * 16, kt * 32);
#pragma unroll
        for (int j = 0; j < 4; ++j)
            acc[j] = wmma_f16(a, ld_b_frag(bsw, (nq * 4 + j) * 8 + kt), acc[j]);
    }
    const int mloc = (lane & 16) ? 8 : 0;
#pragma unroll
    for (int j = 0; j < 4; ++j) {
        int g = (nq * 4 + j) * 16 + (lane & 15);
        float bv = bias_d[g];
        size_t rbase = row0 + mw * 16 + mloc;
#pragma unroll
        for (int v = 0; v < 8; ++v)
            out[(rbase + v) * D_ + g] = acc[j][v] + bv;
    }
}

// ================= Phase 2: sequential recurrence, single WGP =================
#define OFF_WD2H 0          // 16 nt x 8 kt x 1 KB = 128 KB
#define OFF_WD2Z 131072     //  4 nt x 8 kt x 1 KB = 32 KB
#define OFF_WZ2H 163840     // 16 nt x 1 kt x 1 KB = 16 KB
#define OFF_DF16 180224     // 128 x DP halves     = 66 KB
#define OFF_ZF16 247808     // 128 x ZP halves     = 10 KB
#define OFF_DZ   258048     // 128 x DZP floats    = 34 KB
#define OFF_RED  292864     // 1024 floats
#define P2_SMEM  296960     // ~290 KB of the WGP's 320 KB LDS

__global__ __launch_bounds__(1024, 1)
void pvrnn_phase2(const float* __restrict__ Aad, const float* __restrict__ in_p0,
                  const float* __restrict__ noise, const float* __restrict__ Wd2h,
                  const float* __restrict__ Wz2h, const float* __restrict__ Wd2z,
                  const float* __restrict__ init_h, float* __restrict__ out,
                  float* __restrict__ out_kl) {
    extern __shared__ char smem[];
    _Float16* wd2h = (_Float16*)(smem + OFF_WD2H);
    _Float16* wd2z = (_Float16*)(smem + OFF_WD2Z);
    _Float16* wz2h = (_Float16*)(smem + OFF_WZ2H);
    _Float16* dF   = (_Float16*)(smem + OFF_DF16);
    _Float16* zF   = (_Float16*)(smem + OFF_ZF16);
    float*    dzL  = (float*)(smem + OFF_DZ);
    float*    red  = (float*)(smem + OFF_RED);

    const int tid  = threadIdx.x, lane = tid & 31, wave = tid >> 5;
    const int mw   = wave >> 2, nq = wave & 3;      // 8 x 4 wave grid
    const int mloc = (lane & 16) ? 8 : 0;
    const int ncol = lane & 15;

    // ---- stage all recurrent weights into LDS (f32 -> f16, pre-swizzled) ----
    for (int e = tid; e < D_ * D_; e += 1024) {         // B[k][n] = Wd2h[n][k]
        int n = e >> 8, k = e & 255;
        store_b_swz(wd2h, (n >> 4) * 8 + (k >> 5), k & 31, n & 15, Wd2h[e]);
    }
    for (int e = tid; e < D_ * Z_; e += 1024) {         // B[k][n] = Wz2h[n][k], K=32
        int n = e >> 5, k = e & 31;
        store_b_swz(wz2h, n >> 4, k, n & 15, Wz2h[e]);
    }
    for (int e = tid; e < 2 * Z_ * D_; e += 1024) {     // B[k][n] = Wd2z[n][k]
        int n = e >> 8, k = e & 255;
        store_b_swz(wd2z, (n >> 4) * 8 + (k >> 5), k & 31, n & 15, Wd2z[e]);
    }

    // ---- h0 = init_h (C-layout registers), d0 = tanh(h0) into dF ----
    float h[4][8];
#pragma unroll
    for (int j = 0; j < 4; ++j) {
        int g = nq * 64 + j * 16 + ncol;
#pragma unroll
        for (int v = 0; v < 8; ++v) {
            int row = mw * 16 + mloc + v;
            float hv = init_h[row * D_ + g];
            h[j][v] = hv;
            dF[row * DP + g] = (_Float16)tanhf(hv);
        }
    }
    float klacc = 0.0f;
    __syncthreads();

    for (int t = 0; t < T_; ++t) {
        // ---- issue this step's X loads first: hidden behind the GEMM chain ----
        float x[4][8];
#pragma unroll
        for (int j = 0; j < 4; ++j) {
            int g = nq * 64 + j * 16 + ncol;
#pragma unroll
            for (int v = 0; v < 8; ++v) {
                size_t o = (size_t)(mw * 16 + mloc + v) * (T_ * D_) + (size_t)t * D_ + g;
                x[j][v] = out[o];
            }
        }

        // ---- fused GEMM: dz = d@Wd2z^T and pre_rec(Wd2h part) = d@Wd2h^T ----
        // shares one A-fragment load per kt between both products
        v8f az = {};
        v8f acc[4] = {};
#pragma unroll
        for (int kt = 0; kt < 8; ++kt) {
            v16h a = ld_a_frag(dF, DP, mw * 16, kt * 32);
            az = wmma_f16(a, ld_b_frag(wd2z, nq * 8 + kt), az);
#pragma unroll
            for (int j = 0; j < 4; ++j)
                acc[j] = wmma_f16(a, ld_b_frag(wd2h, (nq * 4 + j) * 8 + kt), acc[j]);
        }
#pragma unroll
        for (int v = 0; v < 8; ++v)
            dzL[(mw * 16 + mloc + v) * DZP + nq * 16 + ncol] = az[v];
        __syncthreads();                       // barrier 1: dzL ready, dF reads done

        // ---- posterior sample z, prior stats, KL accumulation ----
        const float wt = ((t == 0) ? W1_ : W_) * (1.0f / (float)Z_);
#pragma unroll
        for (int i = 0; i < 4; ++i) {
            int idx = tid * 4 + i;
            int b = idx >> 5, j = idx & 31;
            size_t base = (size_t)b * T_ + t;
            float a1 = Aad[base * 64 + j];
            float a2 = Aad[base * 64 + 32 + j];
            float nz = noise[base * 32 + j];
            float p1, p2;
            if (t == 0) { p1 = in_p0[b * 64 + j]; p2 = in_p0[b * 64 + 32 + j]; }
            else        { p1 = dzL[b * DZP + j];  p2 = dzL[b * DZP + 32 + j];  }
            float mu_q = tanhf(a1), sq = expf(a2);
            float mu_p = tanhf(p1), sp = expf(p2);
            zF[b * ZP + j] = (_Float16)(mu_q + nz * sq);
            float dmu = mu_q - mu_p;   // mu_q^2 + mu_p^2 - 2 mu_q mu_p
            klacc += wt * (logf(sp + EPS_) - logf(sq + EPS_) - 0.5f
                           + 0.5f * (dmu * dmu + sq * sq) / (sp * sp + EPS_));
            // warm L2/WGP$ for next step's adaptive vars (speculative: safe at t=T-1)
            __builtin_prefetch(&Aad[base * 64 + 64], 0, 3);
            __builtin_prefetch(&noise[base * 32 + 32], 0, 3);
        }
        __syncthreads();                       // barrier 2: zF ready

        // ---- pre_rec += z @ Wz2h^T ----
        {
            v16h a = ld_a_frag(zF, ZP, mw * 16, 0);
#pragma unroll
            for (int j = 0; j < 4; ++j)
                acc[j] = wmma_f16(a, ld_b_frag(wz2h, nq * 4 + j), acc[j]);
        }

        // ---- h = decay*h + (pre_rec + X[t]) * inv_tau ; d = tanh(h) ----
#pragma unroll
        for (int j = 0; j < 4; ++j) {
            int g = nq * 64 + j * 16 + ncol;
            float invt = (g < (D_ / 2)) ? 0.25f : 0.5f;   // tau = 4 / 2
            float dec  = 1.0f - invt;
#pragma unroll
            for (int v = 0; v < 8; ++v) {
                int row = mw * 16 + mloc + v;
                size_t o = (size_t)row * (T_ * D_) + (size_t)t * D_ + g;
                float hn = dec * h[j][v] + (acc[j][v] + x[j][v]) * invt;
                h[j][v] = hn;
                float dv = tanhf(hn);
                out[o] = dv;                    // overwrite X[t] with d[t]
                dF[row * DP + g] = (_Float16)dv;
            }
        }
        __syncthreads();                       // barrier 3: dF[t] published
    }

    // ---- block-reduce weighted KL sum ----
    red[tid] = klacc;
    __syncthreads();
    for (int s = 512; s > 0; s >>= 1) {
        if (tid < s) red[tid] += red[tid + s];
        __syncthreads();
    }
    if (tid == 0) out_kl[0] = red[0];
}

// ================= init-h NLL scalar =================
__global__ void pvrnn_nll(const float* __restrict__ init_h, const float* __restrict__ mu,
                          float* __restrict__ out1) {
    __shared__ float red[256];
    float s = 0.0f;
    for (int i = threadIdx.x; i < B_ * D_; i += 256) {
        float d = init_h[i] - mu[i & (D_ - 1)];
        s += d * d;
    }
    red[threadIdx.x] = s;
    __syncthreads();
    for (int k = 128; k > 0; k >>= 1) {
        if (threadIdx.x < k) red[threadIdx.x] += red[threadIdx.x + k];
        __syncthreads();
    }
    if (threadIdx.x == 0) {
        const float LOG2PI = 1.8378770664093454f;
        float total = (float)(B_ * D_) * 0.5f * LOG2PI + 0.5f * red[0];
        out1[0] = WD1_ * total / (float)D_;
    }
}

// ================= host launch =================
extern "C" void kernel_launch(void* const* d_in, const int* in_sizes, int n_in,
                              void* d_out, int out_size, void* d_ws, size_t ws_size,
                              hipStream_t stream) {
    (void)in_sizes; (void)n_in; (void)out_size; (void)d_ws; (void)ws_size;
    const float* hd     = (const float*)d_in[0];
    const float* A      = (const float*)d_in[1];
    const float* in_p0  = (const float*)d_in[2];
    const float* noise  = (const float*)d_in[3];
    const float* Wd2h   = (const float*)d_in[4];
    const float* Wz2h   = (const float*)d_in[5];
    const float* Wd2z   = (const float*)d_in[6];
    const float* Whd2h  = (const float*)d_in[7];
    const float* bias_d = (const float*)d_in[8];
    const float* init_h = (const float*)d_in[9];
    const float* ih_mu  = (const float*)d_in[10];
    float* out = (float*)d_out;
    const size_t BTD = (size_t)B_ * T_ * D_;

    // Phase 1: X = hd @ Whd2h^T + bias, written into the d_seq region of d_out.
    pvrnn_phase1<<<(B_ * T_) / 128, 1024, P1_SMEM, stream>>>(hd, Whd2h, bias_d, out);
    // Phase 2: persistent single-workgroup recurrence (reads X, overwrites with d).
    pvrnn_phase2<<<1, 1024, P2_SMEM, stream>>>(A, in_p0, noise, Wd2h, Wz2h, Wd2z,
                                               init_h, out, out + BTD);
    // Scalar init-h NLL.
    pvrnn_nll<<<1, 256, 0, stream>>>(init_h, ih_mu, out + BTD + 1);
}
